// NewGAT_78735340470661
// MI455X (gfx1250) — matrix operand order repaired and checked by reading, not compiled, not run on hardware
//
#include <hip/hip_runtime.h>
#include <hip/hip_bf16.h>

// ---------------- constants ----------------
#define FDIM 128
#define HEADS 8
#define CDIM 16
#define NEG_SLOPE 0.2f
#define LN_EPS 1e-6f

typedef __attribute__((ext_vector_type(16))) __bf16 v16bf;
typedef __attribute__((ext_vector_type(8)))  float  v8f;

// ---------------- helpers ----------------
// monotone encoding of float for unsigned atomicMax (0 == -inf identity)
__device__ __forceinline__ unsigned enc_f32(float f) {
    unsigned u = __float_as_uint(f);
    return (u & 0x80000000u) ? ~u : (u | 0x80000000u);
}
__device__ __forceinline__ float dec_f32(unsigned e) {
    unsigned u = (e & 0x80000000u) ? (e & 0x7fffffffu) : ~e;
    return __uint_as_float(u);
}

__device__ __forceinline__ v8f wmma_bf16(v16bf a, v16bf b, v8f c) {
    // v_wmma_f32_16x16x32_bf16  (neg_a, A, neg_b, B, c_mod, C, reuse_a, reuse_b)
    return __builtin_amdgcn_wmma_f32_16x16x32_bf16(false, a, false, b, (short)0, c,
                                                   false, false);
}

// fp32-accurate product via bf16 split: (ahi+alo)(bhi+blo) ~= ahi*bhi + ahi*blo + alo*bhi
__device__ __forceinline__ v8f mm_hl(const v16bf& ah, const v16bf& al,
                                     const v16bf& bh, const v16bf& bl, v8f acc) {
    acc = wmma_bf16(ah, bh, acc);
    acc = wmma_bf16(ah, bl, acc);
    acc = wmma_bf16(al, bh, acc);
    return acc;
}

// Load one 16x32 A fragment (row-major fp32 source, ld = 128) as bf16 hi/lo.
// ISA layout: lanes 0-15 hold M=lane, K = {0..7,16..23}; lanes 16-31: K = {8..15,24..31}.
__device__ __forceinline__ void load_afrag_f32(const float* __restrict__ row, int kbase,
                                               int lane, v16bf& hi, v16bf& lo) {
    const int bk = (lane < 16) ? 0 : 8;
#pragma unroll
    for (int g = 0; g < 2; ++g) {
        const float* p = row + kbase + g * 16 + bk;
#pragma unroll
        for (int i = 0; i < 8; ++i) {
            float v  = p[i];
            __bf16 h = (__bf16)v;
            hi[g * 8 + i] = h;
            lo[g * 8 + i] = (__bf16)(v - (float)h);
        }
    }
}

// ---------------- weight repack: fp32 [K=128][N=128] -> bf16 hi/lo B-fragments --------
// flat index t = ((nt*4 + kc)*32 + lane)*16 + e ; n = nt*16 + (lane&15),
// k = kc*32 + (e&8)*2 + (lane<16 ? 0 : 8) + (e&7)   (matches 32x16 bf16 B-operand layout)
__global__ void k_pack_weight(const float* __restrict__ W,
                              __bf16* __restrict__ hi, __bf16* __restrict__ lo) {
    int t = blockIdx.x * blockDim.x + threadIdx.x;
    if (t >= FDIM * FDIM) return;
    int e    = t & 15;
    int lane = (t >> 4) & 31;
    int kc   = (t >> 9) & 3;
    int nt   = (t >> 11) & 7;
    int n = nt * 16 + (lane & 15);
    int k = kc * 32 + (e & 8) * 2 + ((lane < 16) ? 0 : 8) + (e & 7);
    float v  = W[k * FDIM + n];
    __bf16 h = (__bf16)v;
    hi[t] = h;
    lo[t] = (__bf16)(v - (float)h);
}

// ---------------- zero-init of softmax state ----------------
__global__ void k_init(unsigned* __restrict__ segmax, float* __restrict__ denom,
                       float* __restrict__ agg, int Nn) {
    long long t = (long long)blockIdx.x * blockDim.x + threadIdx.x;
    long long total = (long long)Nn * FDIM;
    if (t >= total) return;
    agg[t] = 0.0f;
    if (t < (long long)Nn * HEADS) {
        segmax[t] = 0u;   // decodes below every finite logit; self-loops guarantee an update
        denom[t]  = 0.0f;
    }
}

// ---------------- projection: x_l = x@W_l + b_l ; x_r = x@W_r + b_r ----------------
__global__ __launch_bounds__(256) void k_proj(
    const float* __restrict__ x,
    const v16bf* __restrict__ wlhi, const v16bf* __restrict__ wllo,
    const v16bf* __restrict__ wrhi, const v16bf* __restrict__ wrlo,
    const float* __restrict__ b_l, const float* __restrict__ b_r,
    float* __restrict__ xl, float* __restrict__ xr, int Nn) {
    const int lane  = threadIdx.x & 31;
    const int wave  = threadIdx.x >> 5;
    const int strip = blockIdx.x * 8 + wave;
    const int rb    = strip * 16;
    if (rb >= Nn) return;                       // wave-uniform exit (EXEC stays full)

    int m = rb + (lane & 15);
    if (m >= Nn) m = Nn - 1;
    const float* arow = x + (size_t)m * FDIM;

    v16bf ah[4], al[4];
#pragma unroll
    for (int kc = 0; kc < 4; ++kc) load_afrag_f32(arow, kc * 32, lane, ah[kc], al[kc]);

    const int col  = lane & 15;
    const int rofs = (lane < 16) ? 0 : 8;
#pragma unroll
    for (int nt = 0; nt < 8; ++nt) {
        float bvl = b_l[nt * 16 + col];
        float bvr = b_r[nt * 16 + col];
        v8f accl, accr;
#pragma unroll
        for (int r = 0; r < 8; ++r) { accl[r] = bvl; accr[r] = bvr; }
#pragma unroll
        for (int kc = 0; kc < 4; ++kc) {
            int fi = (nt * 4 + kc) * 32 + lane;
            accl = mm_hl(ah[kc], al[kc], wlhi[fi], wllo[fi], accl);
            accr = mm_hl(ah[kc], al[kc], wrhi[fi], wrlo[fi], accr);
        }
#pragma unroll
        for (int r = 0; r < 8; ++r) {
            int row = rb + rofs + r;
            if (row < Nn) {
                xl[(size_t)row * FDIM + nt * 16 + col] = accl[r];
                xr[(size_t)row * FDIM + nt * 16 + col] = accr[r];
            }
        }
    }
}

// ---------------- edge phase ----------------
// one wave per edge; lane covers channels [4*lane, 4*lane+4) -> head = lane>>2
__device__ __forceinline__ float edge_logit(const float4 ml, const float4 mr,
                                            const float4 av) {
    float p = 0.f, v;
    v = ml.x + mr.x; v = v > 0.f ? v : NEG_SLOPE * v; p += v * av.x;
    v = ml.y + mr.y; v = v > 0.f ? v : NEG_SLOPE * v; p += v * av.y;
    v = ml.z + mr.z; v = v > 0.f ? v : NEG_SLOPE * v; p += v * av.z;
    v = ml.w + mr.w; v = v > 0.f ? v : NEG_SLOPE * v; p += v * av.w;
    p += __shfl_xor(p, 1, 32);   // butterfly within 4-lane head group
    p += __shfl_xor(p, 2, 32);
    return p;                    // all 4 lanes of the head hold the full logit
}

__device__ __forceinline__ void edge_decode(long long e, const long long* ei,
                                            int Ecnt, long long& s, long long& d) {
    if (e < Ecnt) { s = ei[e]; d = ei[(long long)Ecnt + e]; }
    else          { s = d = e - Ecnt; }        // appended self loops
}

__global__ __launch_bounds__(256) void k_edge_max(
    const float* __restrict__ xl, const float* __restrict__ xr,
    const long long* __restrict__ ei, const float* __restrict__ att,
    unsigned* __restrict__ segmax, int Ecnt, int Nn) {
    long long e = (long long)blockIdx.x * 8 + (threadIdx.x >> 5);
    const int lane = threadIdx.x & 31;
    if (e >= (long long)Ecnt + Nn) return;
    long long s, d;
    edge_decode(e, ei, Ecnt, s, d);
    const float4 ml = *(const float4*)(xl + (size_t)s * FDIM + lane * 4);
    const float4 mr = *(const float4*)(xr + (size_t)d * FDIM + lane * 4);
    const float4 av = *(const float4*)(att + lane * 4);
    float p = edge_logit(ml, mr, av);
    if ((lane & 3) == 0)
        atomicMax(&segmax[(size_t)d * HEADS + (lane >> 2)], enc_f32(p));
}

__global__ __launch_bounds__(256) void k_edge_agg(
    const float* __restrict__ xl, const float* __restrict__ xr,
    const long long* __restrict__ ei, const float* __restrict__ att,
    const unsigned* __restrict__ segmax, float* __restrict__ denom,
    float* __restrict__ agg, int Ecnt, int Nn) {
    long long e = (long long)blockIdx.x * 8 + (threadIdx.x >> 5);
    const int lane = threadIdx.x & 31;
    if (e >= (long long)Ecnt + Nn) return;
    long long s, d;
    edge_decode(e, ei, Ecnt, s, d);
    const float4 ml = *(const float4*)(xl + (size_t)s * FDIM + lane * 4);
    const float4 mr = *(const float4*)(xr + (size_t)d * FDIM + lane * 4);
    const float4 av = *(const float4*)(att + lane * 4);
    float p  = edge_logit(ml, mr, av);
    float mx = dec_f32(segmax[(size_t)d * HEADS + (lane >> 2)]);
    float w  = __expf(p - mx);
    if ((lane & 3) == 0)
        atomicAdd(&denom[(size_t)d * HEADS + (lane >> 2)], w);
    float* ag = agg + (size_t)d * FDIM + lane * 4;   // unnormalized: divide at node level
    atomicAdd(ag + 0, w * ml.x);
    atomicAdd(ag + 1, w * ml.y);
    atomicAdd(ag + 2, w * ml.z);
    atomicAdd(ag + 3, w * ml.w);
}

// ---------------- node finalize: out = agg/denom + bias (in place) ----------------
__global__ void k_finalize(float* __restrict__ agg, const float* __restrict__ denom,
                           const float* __restrict__ bias, int Nn) {
    long long t = (long long)blockIdx.x * blockDim.x + threadIdx.x;
    if (t >= (long long)Nn * FDIM) return;
    int hc = (int)(t & (FDIM - 1));
    long long n = t >> 7;
    agg[t] = agg[t] / denom[n * HEADS + (hc >> 4)] + bias[hc];
}

// ---------------- fused FFN + residual + LayerNorm ----------------
// 4 waves / block (LDS: 4 * 16 * 132 * 4 B = 33 KB, conflict-padded)
__global__ __launch_bounds__(128) void k_ffn(
    const float* __restrict__ outb,
    const v16bf* __restrict__ w1hi, const v16bf* __restrict__ w1lo,
    const v16bf* __restrict__ w2hi, const v16bf* __restrict__ w2lo,
    const float* __restrict__ b1, const float* __restrict__ b2,
    const float* __restrict__ lng, const float* __restrict__ lnb,
    float* __restrict__ y, int Nn) {
    __shared__ float lds[4][16][FDIM + 4];
    const int lane = threadIdx.x & 31;
    const int wave = threadIdx.x >> 5;
    const int nstrips = (Nn + 15) / 16;
    int strip = blockIdx.x * 4 + wave;
    if (strip >= nstrips) strip = nstrips - 1;  // clamp: duplicate work, identical writes
    const int rb   = strip * 16;
    int m = rb + (lane & 15);
    if (m >= Nn) m = Nn - 1;
    const int col  = lane & 15;
    const int rofs = (lane < 16) ? 0 : 8;

    v16bf ah[4], al[4];
#pragma unroll
    for (int kc = 0; kc < 4; ++kc)
        load_afrag_f32(outb + (size_t)m * FDIM, kc * 32, lane, ah[kc], al[kc]);

    // GEMM1 + bias + ReLU -> LDS (wave-private strip)
#pragma unroll
    for (int nt = 0; nt < 8; ++nt) {
        float bv = b1[nt * 16 + col];
        v8f acc;
#pragma unroll
        for (int r = 0; r < 8; ++r) acc[r] = bv;
#pragma unroll
        for (int kc = 0; kc < 4; ++kc) {
            int fi = (nt * 4 + kc) * 32 + lane;
            acc = mm_hl(ah[kc], al[kc], w1hi[fi], w1lo[fi], acc);
        }
#pragma unroll
        for (int r = 0; r < 8; ++r)
            lds[wave][rofs + r][nt * 16 + col] = acc[r] > 0.f ? acc[r] : 0.f;
    }
    __syncthreads();

    // reload A fragments of hidden activations from LDS
#pragma unroll
    for (int kc = 0; kc < 4; ++kc)
        load_afrag_f32(&lds[wave][lane & 15][0], kc * 32, lane, ah[kc], al[kc]);

    // GEMM2 + bias + residual; accumulate LayerNorm stats per row
    v8f hv[8];
    float rsum[8], rsq[8];
#pragma unroll
    for (int r = 0; r < 8; ++r) { rsum[r] = 0.f; rsq[r] = 0.f; }
#pragma unroll
    for (int nt = 0; nt < 8; ++nt) {
        float bv = b2[nt * 16 + col];
        v8f acc;
#pragma unroll
        for (int r = 0; r < 8; ++r) acc[r] = bv;
#pragma unroll
        for (int kc = 0; kc < 4; ++kc) {
            int fi = (nt * 4 + kc) * 32 + lane;
            acc = mm_hl(ah[kc], al[kc], w2hi[fi], w2lo[fi], acc);
        }
#pragma unroll
        for (int r = 0; r < 8; ++r) {
            int row = rb + rofs + r;
            int rr  = row < Nn ? row : Nn - 1;
            float h = acc[r] + outb[(size_t)rr * FDIM + nt * 16 + col];
            hv[nt][r] = h;
            rsum[r] += h;
            rsq[r]  += h * h;
        }
    }
    // cross-lane reduction over the 16 lanes sharing each output row
#pragma unroll
    for (int r = 0; r < 8; ++r) {
        float s = rsum[r], q = rsq[r];
#pragma unroll
        for (int msk = 1; msk < 16; msk <<= 1) {
            s += __shfl_xor(s, msk, 16);
            q += __shfl_xor(q, msk, 16);
        }
        rsum[r] = s; rsq[r] = q;
    }
    // LayerNorm + affine + store
#pragma unroll
    for (int nt = 0; nt < 8; ++nt) {
        float gg = lng[nt * 16 + col];
        float bb = lnb[nt * 16 + col];
#pragma unroll
        for (int r = 0; r < 8; ++r) {
            int row = rb + rofs + r;
            if (row < Nn) {
                float mu  = rsum[r] * (1.0f / FDIM);
                float var = rsq[r] * (1.0f / FDIM) - mu * mu;
                float rs  = rsqrtf(var + LN_EPS);
                y[(size_t)row * FDIM + nt * 16 + col] = (hv[nt][r] - mu) * rs * gg + bb;
            }
        }
    }
}

// ---------------- host side ----------------
extern "C" void kernel_launch(void* const* d_in, const int* in_sizes, int n_in,
                              void* d_out, int out_size, void* d_ws, size_t ws_size,
                              hipStream_t stream) {
    const float*     x    = (const float*)d_in[0];
    const long long* ei   = (const long long*)d_in[1];
    const float*     W_l  = (const float*)d_in[2];
    const float*     b_l  = (const float*)d_in[3];
    const float*     W_r  = (const float*)d_in[4];
    const float*     b_r  = (const float*)d_in[5];
    const float*     att  = (const float*)d_in[6];
    const float*     bias = (const float*)d_in[7];
    const float*     w1   = (const float*)d_in[8];
    const float*     b1   = (const float*)d_in[9];
    const float*     w2   = (const float*)d_in[10];
    const float*     b2   = (const float*)d_in[11];
    const float*     lng  = (const float*)d_in[12];
    const float*     lnb  = (const float*)d_in[13];

    const int Nn   = in_sizes[0] / FDIM;
    const int Ecnt = in_sizes[1] / 2;

    // ---- carve workspace ----
    char*  ws  = (char*)d_ws;
    size_t off = 0;
    auto carve = [&](size_t bytes) -> void* {
        void* p = ws + off;
        off += (bytes + 255) & ~(size_t)255;
        return p;
    };
    const size_t wbytes = (size_t)FDIM * FDIM * sizeof(__bf16);  // 32 KB per half
    __bf16* wlhi = (__bf16*)carve(wbytes);
    __bf16* wllo = (__bf16*)carve(wbytes);
    __bf16* wrhi = (__bf16*)carve(wbytes);
    __bf16* wrlo = (__bf16*)carve(wbytes);
    __bf16* w1hi = (__bf16*)carve(wbytes);
    __bf16* w1lo = (__bf16*)carve(wbytes);
    __bf16* w2hi = (__bf16*)carve(wbytes);
    __bf16* w2lo = (__bf16*)carve(wbytes);
    float*    xl     = (float*)carve((size_t)Nn * FDIM * sizeof(float));
    float*    xr     = (float*)carve((size_t)Nn * FDIM * sizeof(float));
    unsigned* segmax = (unsigned*)carve((size_t)Nn * HEADS * sizeof(unsigned));
    float*    denom  = (float*)carve((size_t)Nn * HEADS * sizeof(float));
    float*    agg    = (float*)carve((size_t)Nn * FDIM * sizeof(float));
    (void)ws_size; (void)n_in; (void)out_size;

    // 1) repack weights into WMMA B-fragment order (bf16 hi/lo)
    {
        dim3 g((FDIM * FDIM + 255) / 256), b(256);
        k_pack_weight<<<g, b, 0, stream>>>(W_l, wlhi, wllo);
        k_pack_weight<<<g, b, 0, stream>>>(W_r, wrhi, wrlo);
        k_pack_weight<<<g, b, 0, stream>>>(w1, w1hi, w1lo);
        k_pack_weight<<<g, b, 0, stream>>>(w2, w2hi, w2lo);
    }
    // 2) zero softmax state
    {
        long long total = (long long)Nn * FDIM;
        k_init<<<(unsigned)((total + 255) / 256), 256, 0, stream>>>(segmax, denom, agg, Nn);
    }
    // 3) projections (WMMA)
    const int nstrips = (Nn + 15) / 16;
    k_proj<<<(nstrips + 7) / 8, 256, 0, stream>>>(
        x, (const v16bf*)wlhi, (const v16bf*)wllo, (const v16bf*)wrhi, (const v16bf*)wrlo,
        b_l, b_r, xl, xr, Nn);
    // 4) + 5) edge phase (wave per edge, L2-resident gathers)
    {
        long long Et = (long long)Ecnt + Nn;
        unsigned grid = (unsigned)((Et + 7) / 8);
        k_edge_max<<<grid, 256, 0, stream>>>(xl, xr, ei, att, segmax, Ecnt, Nn);
        k_edge_agg<<<grid, 256, 0, stream>>>(xl, xr, ei, att, segmax, denom, agg, Ecnt, Nn);
    }
    // 6) node-level normalization + bias (in place)
    {
        long long total = (long long)Nn * FDIM;
        k_finalize<<<(unsigned)((total + 255) / 256), 256, 0, stream>>>(agg, denom, bias, Nn);
    }
    // 7) fused FFN + residual + LayerNorm (WMMA) -> d_out
    k_ffn<<<(nstrips + 3) / 4, 128, 0, stream>>>(
        agg, (const v16bf*)w1hi, (const v16bf*)w1lo, (const v16bf*)w2hi, (const v16bf*)w2lo,
        b1, b2, lng, lnb, (float*)d_out, Nn);
}
